// GAT_36292473651369
// MI455X (gfx1250) — compile-verified, matrix operand order checked by reading, not run on hardware
//
#include <hip/hip_runtime.h>
#include <hip/hip_bf16.h>
#include <math.h>

typedef __attribute__((ext_vector_type(16))) _Float16 v16h;
typedef __attribute__((ext_vector_type(8)))  _Float16 v8h;
typedef __attribute__((ext_vector_type(8)))  float    v8f;

#define HIDDEN 128
#define WAVES  4   // 4 waves / block, each wave owns a 16-row Q tile

__device__ __forceinline__ v8f zero8() {
    v8f z;
#pragma unroll
    for (int i = 0; i < 8; i++) z[i] = 0.0f;
    return z;
}

__device__ __forceinline__ v16h make16(v8h lo, v8h hi) {
    v16h r;
#pragma unroll
    for (int i = 0; i < 8; i++) { r[i] = lo[i]; r[i + 8] = hi[i]; }
    return r;
}

// A-operand (16x32 f16) from a row-major [.][128] matrix.
// Lane layout (ISA 7.12.2): lanes 0-15 row M=lane, halves K = k0+{0..7} then k0+16+{0..7};
// lanes 16-31 same rows, K = k0+8+{0..7} then k0+24+{0..7}.
// The B-operand of QK^T (B = K^T) has the mirrored layout (lane = N), so the
// same loader with rowBase = j0 produces it from row-major K.
__device__ __forceinline__ v16h load_tileA(const _Float16* __restrict__ M, int ld,
                                           int rowBase, int lane, int k0) {
    int row = rowBase + (lane & 15);
    int sel = (lane & 16) ? 8 : 0;
    const _Float16* p = M + (long)row * ld + k0 + sel;
    v8h lo = *(const v8h*)p;
    v8h hi = *(const v8h*)(p + 16);
    return make16(lo, hi);
}

// B-operand (32x16 f16) for PV from the transposed V matrix VhT[feature][row].
// Lane holds feature n = f0 + (lane&15); halves enumerate K (= item rows) as j0+sel+{0..7}, then j0+16+sel+{0..7}.
__device__ __forceinline__ v16h load_tileBT(const _Float16* __restrict__ MT, int ld,
                                            int f0, int lane, int j0) {
    int f = f0 + (lane & 15);
    int sel = (lane & 16) ? 8 : 0;
    const _Float16* p = MT + (long)f * ld + j0 + sel;
    v8h lo = *(const v8h*)p;
    v8h hi = *(const v8h*)(p + 16);
    return make16(lo, hi);
}

// ---------------------------------------------------------------------------
// Projection: Y = concat(relu(X@W1+b1), relu(X@W2+b2)); emits f32, f16, f16^T
// ---------------------------------------------------------------------------
__global__ __launch_bounds__(128) void proj_kernel(
    const float* __restrict__ X,
    const float* __restrict__ W1, const float* __restrict__ b1,
    const float* __restrict__ W2, const float* __restrict__ b2,
    float* __restrict__ Yf, _Float16* __restrict__ Yh,
    _Float16* __restrict__ YhT, int N)
{
    __shared__ float xs[8][HIDDEN];
    const int t  = threadIdx.x;
    const int r0 = blockIdx.x * 8;

#pragma unroll
    for (int i = 0; i < 8; i++) {
        int idx = i * 128 + t;
        xs[idx >> 7][idx & 127] = X[(long)r0 * HIDDEN + idx];
    }
    __syncthreads();

    const float* W = (t < 64) ? W1 : W2;
    const float* b = (t < 64) ? b1 : b2;
    const int    col  = t & 63;
    const float  bias = b[col];

#pragma unroll 1
    for (int i = 0; i < 8; i++) {
        float s = bias;
#pragma unroll 8
        for (int k = 0; k < HIDDEN; k++) s += xs[i][k] * W[k * 64 + col];
        s = fmaxf(s, 0.0f);
        const int row = r0 + i;
        Yf[(long)row * HIDDEN + t] = s;
        _Float16 h = (_Float16)s;
        Yh[(long)row * HIDDEN + t] = h;
        YhT[(long)t * N + row]     = h;   // transposed copy for the PV B-operand
    }
}

// ---------------------------------------------------------------------------
// Flash-attention style: out = softmax(mask * (Q K^T) / sqrt(H)) @ V + Qbase
// One wave per 16-row Q tile; 32 K/V columns per iteration; online softmax.
// ---------------------------------------------------------------------------
__global__ __launch_bounds__(WAVES * 32) void attn_kernel(
    const _Float16* __restrict__ Qh,   // [NQ][128] f16
    const _Float16* __restrict__ Kh,   // [NK][128] f16
    const _Float16* __restrict__ VhT,  // [128][NK] f16 (transposed V == K matrix)
    const float*    __restrict__ Qbase,// [NQ][128] f32 residual
    const int*      __restrict__ adj,  // [NQ][NK]
    float*          __restrict__ out,  // [NQ][128] f32
    int NQ, int NK)
{
    __shared__ __align__(16) _Float16 pbuf[WAVES][16][32];

    const int lane = threadIdx.x & 31;
    const int wave = threadIdx.x >> 5;
    const int i0   = (blockIdx.x * WAVES + wave) * 16;
    if (i0 >= NQ) return;

    // Q tile in A layout, all 4 K=32 chunks (held in registers for the whole loop)
    v16h a[4];
#pragma unroll
    for (int c = 0; c < 4; c++) a[c] = load_tileA(Qh, HIDDEN, i0, lane, c * 32);

    float m[8], l[8];
    v8f   acc[8];
#pragma unroll
    for (int v = 0; v < 8; v++) { m[v] = -INFINITY; l[v] = 0.0f; }
#pragma unroll
    for (int fc = 0; fc < 8; fc++) acc[fc] = zero8();

    const float scale = 0.08838834764831845f; // 1/sqrt(128)
    _Float16* myp = &pbuf[wave][0][0];
    const int nsel   = lane & 15;
    const int rsel   = (lane & 16) ? 8 : 0;

    for (int j0 = 0; j0 < NK; j0 += 32) {
        // ---- scores: two 16x16 tiles over K=128 ----
        v8f c0 = zero8(), c1 = zero8();
#pragma unroll
        for (int c = 0; c < 4; c++) {
            v16h b0 = load_tileA(Kh, HIDDEN, j0,      lane, c * 32);
            v16h b1 = load_tileA(Kh, HIDDEN, j0 + 16, lane, c * 32);
            c0 = __builtin_amdgcn_wmma_f32_16x16x32_f16(false, a[c], false, b0,
                                                        (short)0, c0, false, false);
            c1 = __builtin_amdgcn_wmma_f32_16x16x32_f16(false, a[c], false, b1,
                                                        (short)0, c1, false, false);
        }

        // ---- mask-by-multiply (NT loads: adjacency streams once, keep L2 clean) ----
        float s0[8], s1[8];
#pragma unroll
        for (int v = 0; v < 8; v++) {
            long r = (long)(i0 + rsel + v) * NK + j0 + nsel;
            int m0 = __builtin_nontemporal_load(adj + r);
            int m1 = __builtin_nontemporal_load(adj + r + 16);
            s0[v] = c0[v] * (float)m0 * scale;
            s1[v] = c1[v] * (float)m1 * scale;
        }

        // ---- online softmax (row reduction = butterfly over each 16-lane half) ----
        float corr[8];
#pragma unroll
        for (int v = 0; v < 8; v++) {
            float t = fmaxf(s0[v], s1[v]);
            t = fmaxf(t, __shfl_xor(t, 1));
            t = fmaxf(t, __shfl_xor(t, 2));
            t = fmaxf(t, __shfl_xor(t, 4));
            t = fmaxf(t, __shfl_xor(t, 8));
            float mn = fmaxf(m[v], t);
            corr[v]  = __expf(m[v] - mn);
            m[v]     = mn;
            float p0 = __expf(s0[v] - mn);
            float p1 = __expf(s1[v] - mn);
            float rs = p0 + p1;
            rs += __shfl_xor(rs, 1);
            rs += __shfl_xor(rs, 2);
            rs += __shfl_xor(rs, 4);
            rs += __shfl_xor(rs, 8);
            l[v] = l[v] * corr[v] + rs;
            s0[v] = p0; s1[v] = p1;
        }

        // ---- P (C/D layout) -> LDS -> A layout for the PV matmul ----
#pragma unroll
        for (int v = 0; v < 8; v++) {
            int row = v + rsel;
            myp[row * 32 + nsel]      = (_Float16)s0[v];
            myp[row * 32 + nsel + 16] = (_Float16)s1[v];
        }
        asm volatile("s_wait_dscnt 0" ::: "memory");

        {
            int sel = (lane & 16) ? 8 : 0;
            const _Float16* pp = myp + nsel * 32 + sel;
            v8h lo = *(const v8h*)pp;        // K = sel .. sel+7
            v8h hi = *(const v8h*)(pp + 16); // K = 16+sel .. 16+sel+7
            v16h pA = make16(lo, hi);

#pragma unroll
            for (int fc = 0; fc < 8; fc++) {
                v8f t = acc[fc];
#pragma unroll
                for (int v = 0; v < 8; v++) t[v] *= corr[v];
                v16h bV = load_tileBT(VhT, NK, fc * 16, lane, j0);
                acc[fc] = __builtin_amdgcn_wmma_f32_16x16x32_f16(false, pA, false, bV,
                                                                 (short)0, t, false, false);
            }
        }
    }

    // ---- epilogue: normalize, residual add, coalesced f32 stores ----
#pragma unroll
    for (int fc = 0; fc < 8; fc++) {
#pragma unroll
        for (int v = 0; v < 8; v++) {
            int row = i0 + rsel + v;
            int f   = fc * 16 + nsel;
            out[(long)row * HIDDEN + f] =
                acc[fc][v] / l[v] + Qbase[(long)row * HIDDEN + f];
        }
    }
}

// ---------------------------------------------------------------------------
extern "C" void kernel_launch(void* const* d_in, const int* in_sizes, int n_in,
                              void* d_out, int out_size, void* d_ws, size_t ws_size,
                              hipStream_t stream)
{
    (void)in_sizes; (void)n_in; (void)out_size; (void)ws_size;

    const float* ufea = (const float*)d_in[0];
    const float* vfea = (const float*)d_in[1];
    const float* W_u1 = (const float*)d_in[2];
    const float* b_u1 = (const float*)d_in[3];
    const float* W_u2 = (const float*)d_in[4];
    const float* b_u2 = (const float*)d_in[5];
    const float* W_v1 = (const float*)d_in[6];
    const float* b_v1 = (const float*)d_in[7];
    const float* W_v2 = (const float*)d_in[8];
    const float* b_v2 = (const float*)d_in[9];
    const int*   UV   = (const int*)d_in[10];
    const int*   VU   = (const int*)d_in[11];

    const int NU = 8000, NV = 8000;

    char* ws = (char*)d_ws;                 // 16,384,000 bytes used, 256B-aligned slabs
    float*    user_f  = (float*)(ws);                      // 4,096,000 B
    float*    item_f  = (float*)(ws + 4096000);            // 4,096,000 B
    _Float16* user_h  = (_Float16*)(ws + 8192000);         // 2,048,000 B
    _Float16* item_h  = (_Float16*)(ws + 10240000);        // 2,048,000 B
    _Float16* user_hT = (_Float16*)(ws + 12288000);        // 2,048,000 B
    _Float16* item_hT = (_Float16*)(ws + 14336000);        // 2,048,000 B

    proj_kernel<<<NU / 8, 128, 0, stream>>>(ufea, W_u1, b_u1, W_u2, b_u2,
                                            user_f, user_h, user_hT, NU);
    proj_kernel<<<NV / 8, 128, 0, stream>>>(vfea, W_v1, b_v1, W_v2, b_v2,
                                            item_f, item_h, item_hT, NV);

    float* out_u = (float*)d_out;
    float* out_v = out_u + (long)NU * HIDDEN;

    const int tilesU = NU / 16, tilesV = NV / 16;
    attn_kernel<<<(tilesU + WAVES - 1) / WAVES, WAVES * 32, 0, stream>>>(
        user_h, item_h, item_hT, user_f, UV, out_u, NU, NV);
    attn_kernel<<<(tilesV + WAVES - 1) / WAVES, WAVES * 32, 0, stream>>>(
        item_h, user_h, user_hT, item_f, VU, out_v, NV, NU);
}
